// BipartiteSchedulerGNN_41979010351696
// MI455X (gfx1250) — matrix operand order repaired and checked by reading, not compiled, not run on hardware
//
#include <hip/hip_runtime.h>
#include <hip/hip_bf16.h>

typedef __attribute__((ext_vector_type(16))) _Float16 v16h;
typedef __attribute__((ext_vector_type(8)))  float    v8f;

#define WMMA_F16(A, B, C) \
    __builtin_amdgcn_wmma_f32_16x16x32_f16(false, (A), false, (B), (short)0, (C), false, false)

// A-fragment K index for element e (0..15) of a 16x32 f16 A tile, contiguous
// layout: lanes 0-15 hold K in {0..15}, lanes 16-31 hold {16..31}.
__device__ __forceinline__ int kpatA(int e, int hl) {
    return hl * 16 + e;
}

// One workgroup per (b,u): 8 waves, wave32.
//   Stage 1: agg[k][m] = sum_a relu( relu(x_{a,k} * We1 + be1) @ We2 + be2 )
//   Stage 2: u = relu(relu(agg@Wu1+bu1)@Wu2+bu2); out[k] = u[k,:]@Wo + bo
__global__ __launch_bounds__(256, 1)
void gnn_fused_kernel(const float* __restrict__ ef,
                      const float* __restrict__ We1, const float* __restrict__ be1,
                      const float* __restrict__ We2, const float* __restrict__ be2,
                      const float* __restrict__ Wu1, const float* __restrict__ bu1,
                      const float* __restrict__ Wu2, const float* __restrict__ bu2,
                      const float* __restrict__ Wo,  const float* __restrict__ bo,
                      float* __restrict__ out)
{
    __shared__ float agg[64 * 64];              // [k][m], fp32 accumulator

    const int tid  = threadIdx.x;
    const int lane = tid & 31;
    const int wave = tid >> 5;
    const int hl   = lane >> 4;                 // half-wave (0/1)
    const int lm   = lane & 15;
    const int bu   = blockIdx.x;                // b*64 + u
    const float* __restrict__ efb = ef + (size_t)bu * (32 * 64);

    // zero the LDS accumulator
    for (int i = tid; i < 64 * 64; i += 256) agg[i] = 0.0f;

    // ---- per-lane We1/be1 constants in A-fragment order, packed f16 ----
    v16h w1h, b1h, w1h2, b1h2;
#pragma unroll
    for (int e = 0; e < 16; ++e) {
        int K0 = kpatA(e, hl);
        w1h[e]  = (_Float16)We1[K0];
        b1h[e]  = (_Float16)be1[K0];
        w1h2[e] = (_Float16)We1[32 + K0];
        b1h2[e] = (_Float16)be1[32 + K0];
    }

    // ---- We2 (64x64) as 4 n-tiles x 2 k-frags of 32x16 f16 B fragments ----
    // B layout: lane -> col N = lane%16; VGPR j holds K = 16*hl + 2j, 2j+1.
    v16h bW[4][2];
#pragma unroll
    for (int nt = 0; nt < 4; ++nt)
#pragma unroll
        for (int kf = 0; kf < 2; ++kf)
#pragma unroll
            for (int e = 0; e < 16; ++e) {
                int K = kf * 32 + hl * 16 + e;
                bW[nt][kf][e] = (_Float16)We2[K * 64 + nt * 16 + lm];
            }

    float be2v[4];
#pragma unroll
    for (int nt = 0; nt < 4; ++nt) be2v[nt] = be2[nt * 16 + lm];

    __syncthreads();                            // LDS zeroing visible

    // ---- Stage 1: wave owns k-tile (wave&3) and a in [ (wave>>2)*16, +16 ) ----
    const int ktile = wave & 3;
    const int a0    = (wave >> 2) * 16;

    float acc[4][8];
#pragma unroll
    for (int nt = 0; nt < 4; ++nt)
#pragma unroll
        for (int j = 0; j < 8; ++j) acc[nt][j] = 0.0f;

    const v16h zh = {};
    // software-pipelined edge-scalar load: fetch ai+1 while WMMAs for ai run
    float xf = efb[a0 * 64 + ktile * 16 + lm];

    for (int ai = 0; ai < 16; ++ai) {
        float xf_next = 0.0f;
        if (ai < 15) xf_next = efb[(a0 + ai + 1) * 64 + ktile * 16 + lm];

        // splat x to packed f16, then h = max(x*w1 + b1, 0) as v_pk_fma/max
        const _Float16 xh = (_Float16)xf;
        v16h xs;
#pragma unroll
        for (int e = 0; e < 16; ++e) xs[e] = xh;

        v16h af[2];
        af[0] = __builtin_elementwise_max((v16h)(xs * w1h  + b1h),  zh);
        af[1] = __builtin_elementwise_max((v16h)(xs * w1h2 + b1h2), zh);

#pragma unroll
        for (int nt = 0; nt < 4; ++nt) {
            v8f c = {};
            c = WMMA_F16(af[0], bW[nt][0], c);
            c = WMMA_F16(af[1], bW[nt][1], c);
            // per-edge bias + relu BEFORE summing over APs
#pragma unroll
            for (int j = 0; j < 8; ++j)
                acc[nt][j] += fmaxf(c[j] + be2v[nt], 0.0f);
        }
        xf = xf_next;
    }

    // ---- cross-wave reduction: ds_add_f32 into agg[k][n] ----
#pragma unroll
    for (int nt = 0; nt < 4; ++nt)
#pragma unroll
        for (int j = 0; j < 8; ++j) {
            int k = ktile * 16 + j + 8 * hl;    // C layout: M = j + 8*hl
            atomicAdd(&agg[k * 64 + nt * 16 + lm], acc[nt][j]);
        }
    __syncthreads();

    // ---- Stage 2: two 64x64 dense+relu layers, in place in LDS ----
    // Wave w (0..3) owns k-rows [16w, 16w+16): it reads and writes only its
    // own rows, and per-wave LDS ordering makes layer1->layer2 safe.
    if (wave < 4) {
        const int mt = wave;
#pragma unroll 1
        for (int layer = 0; layer < 2; ++layer) {
            const float* __restrict__ W  = layer ? Wu2 : Wu1;
            const float* __restrict__ bb = layer ? bu2 : bu1;

            v16h aF[2];
#pragma unroll
            for (int f = 0; f < 2; ++f)
#pragma unroll
                for (int e = 0; e < 16; ++e)
                    aF[f][e] = (_Float16)agg[(mt * 16 + lm) * 64 + f * 32 + kpatA(e, hl)];

            float res[4][8];
#pragma unroll
            for (int nt = 0; nt < 4; ++nt) {
                v16h b0, b1;
#pragma unroll
                for (int e = 0; e < 16; ++e) {
                    b0[e] = (_Float16)W[(hl * 16 + e) * 64 + nt * 16 + lm];
                    b1[e] = (_Float16)W[(32 + hl * 16 + e) * 64 + nt * 16 + lm];
                }
                v8f c = {};
                c = WMMA_F16(aF[0], b0, c);
                c = WMMA_F16(aF[1], b1, c);
                const float bias = bb[nt * 16 + lm];
#pragma unroll
                for (int j = 0; j < 8; ++j)
                    res[nt][j] = fmaxf(c[j] + bias, 0.0f);
            }
            // all reads done (A frags in regs) -> safe in-place store
#pragma unroll
            for (int nt = 0; nt < 4; ++nt)
#pragma unroll
                for (int j = 0; j < 8; ++j)
                    agg[(mt * 16 + j + 8 * hl) * 64 + nt * 16 + lm] = res[nt][j];
        }
    }
    __syncthreads();

    // ---- final 64 -> 1 projection ----
    if (tid < 64) {
        float s = bo[0];
#pragma unroll 8
        for (int h = 0; h < 64; ++h)
            s = fmaf(agg[tid * 64 + h], Wo[h], s);
        out[bu * 64 + tid] = s;
    }
}

extern "C" void kernel_launch(void* const* d_in, const int* in_sizes, int n_in,
                              void* d_out, int out_size, void* d_ws, size_t ws_size,
                              hipStream_t stream) {
    const float* ef  = (const float*)d_in[0];
    const float* We1 = (const float*)d_in[1];
    const float* be1 = (const float*)d_in[2];
    const float* We2 = (const float*)d_in[3];
    const float* be2 = (const float*)d_in[4];
    const float* Wu1 = (const float*)d_in[5];
    const float* bu1 = (const float*)d_in[6];
    const float* Wu2 = (const float*)d_in[7];
    const float* bu2 = (const float*)d_in[8];
    const float* Wo  = (const float*)d_in[9];
    const float* bo  = (const float*)d_in[10];
    float* out = (float*)d_out;

    const int BU = in_sizes[0] / (32 * 64);     // B*U = 512
    gnn_fused_kernel<<<dim3(BU), dim3(256), 0, stream>>>(
        ef, We1, be1, We2, be2, Wu1, bu1, Wu2, bu2, Wo, bo, out);
}
